// DeformableAttention2_4518305595599
// MI455X (gfx1250) — matrix operand order repaired and checked by reading, not compiled, not run on hardware
//
#include <hip/hip_runtime.h>

#define B_SZ   8
#define C_DIM  512
#define N_DIM  1024
#define HEADS  8
#define HD     64
#define KSTEP  32
#define KPAD   8                      // LDS row stride 40 shorts = 80B (16B-aligned chunks)
#define ASTRIDE (KSTEP + KPAD)
#define NKT    (C_DIM / KSTEP)        // 16 K-tiles

typedef __bf16 v16bf __attribute__((ext_vector_type(16)));
typedef float  v8f   __attribute__((ext_vector_type(8)));

union Frag { v16bf v; unsigned int u[8]; unsigned short h[16]; };
union Acc  { v8f v; float f[8]; };

__device__ __forceinline__ unsigned short f2bf(float f) {
  union { float f; unsigned int u; } c; c.f = f;
  unsigned int u = c.u;
  u += 0x7FFFu + ((u >> 16) & 1u);          // round-to-nearest-even
  return (unsigned short)(u >> 16);
}

__device__ __forceinline__ v8f wmma_bf16(const Frag& a, const Frag& b, v8f c) {
  return __builtin_amdgcn_wmma_f32_16x16x32_bf16(false, a.v, false, b.v,
                                                 (short)0, c, false, false);
}

// low 32 bits of a generic shared pointer == LDS byte offset (aperture rule, ISA 10.2)
__device__ __forceinline__ unsigned lds_off(const void* p) {
  return (unsigned)(unsigned long long)p;
}

// async global->LDS 16B copy, tracked by ASYNCcnt (ISA 15.18.3 op 98)
__device__ __forceinline__ void async_cp16(unsigned lds, const void* g) {
  asm volatile("global_load_async_to_lds_b128 %0, %1, off"
               :: "v"(lds), "v"((unsigned long long)g) : "memory");
}
__device__ __forceinline__ void wait_async0() {
  asm volatile("s_wait_asynccnt 0x0" ::: "memory");
}

// ---------------------------------------------------------------------------
// One-time fp32 -> bf16 conversions
// ---------------------------------------------------------------------------
__global__ __launch_bounds__(256) void cvt_kernel(
    const float* __restrict__ src, unsigned short* __restrict__ dst) {
  size_t i = (size_t)blockIdx.x * 256 + threadIdx.x;
  dst[i] = f2bf(src[i]);
}

// x[b,c,n] (f32) -> xT[b,n,c] (bf16): K-major for async GEMM staging
__global__ __launch_bounds__(256) void cvt_xT_kernel(
    const float* __restrict__ x, unsigned short* __restrict__ xT) {
  size_t idx = (size_t)blockIdx.x * 256 + threadIdx.x;
  int b = (int)(idx / ((size_t)C_DIM * N_DIM));
  int r = (int)(idx % ((size_t)C_DIM * N_DIM));
  int c = r / N_DIM, n = r % N_DIM;
  xT[((size_t)b * N_DIM + n) * C_DIM + c] = f2bf(x[idx]);
}

// ---------------------------------------------------------------------------
// Selection mask: logits = W_s @ x + b_s (R=4), gumbel noise, argmax==0
// ---------------------------------------------------------------------------
__global__ __launch_bounds__(256) void mask_kernel(
    const float* __restrict__ x, const float* __restrict__ ws,
    const float* __restrict__ bs, const float* __restrict__ u,
    int* __restrict__ mask) {
  int idx = blockIdx.x * 256 + threadIdx.x;
  int b = idx / N_DIM, n = idx % N_DIM;
  const float* xb = x + (size_t)b * C_DIM * N_DIM + n;
  float acc[4] = {0.f, 0.f, 0.f, 0.f};
  for (int c = 0; c < C_DIM; ++c) {
    float xv = xb[(size_t)c * N_DIM];
    acc[0] += xv * ws[0 * C_DIM + c];
    acc[1] += xv * ws[1 * C_DIM + c];
    acc[2] += xv * ws[2 * C_DIM + c];
    acc[3] += xv * ws[3 * C_DIM + c];
  }
  float best = -3.4e38f; int bestr = 0;
  for (int r = 0; r < 4; ++r) {
    float uu = u[((size_t)b * 4 + r) * N_DIM + n];
    float g  = -__logf(-__logf(uu + 1e-10f) + 1e-10f);
    float v  = acc[r] + bs[r] + g;
    if (v > best) { best = v; bestr = r; }
  }
  mask[idx] = (bestr == 0) ? 1 : 0;
}

// ---------------------------------------------------------------------------
// Batched GEMM, bf16 inputs, double-buffered async LDS staging.
//   Out[b,o,n] = postscale * (sum_c W[o,c] * XT[b,n,c] + bias[o])
// MODE 0: store f32 [b][o][n]   MODE 1: store bf16 transposed [b][n][o]
// block = 256 threads (8 waves), tile 128(O) x 64(N), K-step 32.
// ---------------------------------------------------------------------------
template <int MODE>
__global__ __launch_bounds__(256) void gemm512b(
    const unsigned short* __restrict__ XT,   // [B][N][C] bf16
    const unsigned short* __restrict__ Wb,   // [O][C]    bf16
    const float* __restrict__ bias, void* __restrict__ Out, float postscale) {
  alignas(16) __shared__ unsigned short Alds[2][128][ASTRIDE];   // [buf][o][k]
  alignas(16) __shared__ unsigned short Blds[2][64][ASTRIDE];    // [buf][n][k]
  const int tid  = threadIdx.x;
  const int lane = tid & 31, wave = tid >> 5;
  const int b  = blockIdx.z;
  const int o0 = blockIdx.x * 128;
  const int n0 = blockIdx.y * 64;
  const unsigned short* Xb = XT + (size_t)b * N_DIM * C_DIM;
  const int wo = (wave >> 1) * 32;
  const int wn = (wave & 1) * 32;
  const int m  = lane & 15, kh = lane >> 4;

  const int rA = tid >> 2, sA = (tid & 3) * 8;   // A: 2 chunks/thread (rows rA, rA+64)
  const int rB = tid >> 2, sB = (tid & 3) * 8;   // B: 1 chunk/thread

  auto stage = [&](int kc, int buf) {
    async_cp16(lds_off(&Alds[buf][rA][sA]),
               Wb + (size_t)(o0 + rA) * C_DIM + kc + sA);
    async_cp16(lds_off(&Alds[buf][rA + 64][sA]),
               Wb + (size_t)(o0 + 64 + rA) * C_DIM + kc + sA);
    async_cp16(lds_off(&Blds[buf][rB][sB]),
               Xb + (size_t)(n0 + rB) * C_DIM + kc + sB);
    if (kc + KSTEP < C_DIM) {   // prefetch the tile after next into GL2
      __builtin_prefetch(Wb + (size_t)(o0 + rA) * C_DIM + kc + KSTEP + sA, 0, 0);
      __builtin_prefetch(Xb + (size_t)(n0 + rB) * C_DIM + kc + KSTEP + sB, 0, 0);
    }
  };

  Acc acc[2][2];
  for (int i = 0; i < 2; ++i)
    for (int j = 0; j < 2; ++j)
      for (int e = 0; e < 8; ++e) acc[i][j].f[e] = 0.f;

  stage(0, 0);                                   // prologue
  for (int t = 0; t < NKT; ++t) {
    const int buf = t & 1;
    wait_async0();            // this wave's copies for tile t are in LDS
    __syncthreads();          // all waves' copies visible; prior reads of buf^1 retired
    if (t + 1 < NKT) stage((t + 1) * KSTEP, buf ^ 1);   // overlap DMA with compute

    // ---- fragments (ISA 7.12.2 layouts) ----
    Frag a[2], bf[2];
    #pragma unroll
    for (int ti = 0; ti < 2; ++ti) {
      const unsigned short* row = &Alds[buf][wo + ti * 16 + m][0];
      #pragma unroll
      for (int j = 0; j < 4; ++j) {
        a[ti].u[j]     = *(const unsigned int*)&row[kh * 8 + 2 * j];
        a[ti].u[4 + j] = *(const unsigned int*)&row[16 + kh * 8 + 2 * j];
      }
    }
    const int koff = kh * 16;
    #pragma unroll
    for (int tj = 0; tj < 2; ++tj) {
      const unsigned short* row = &Blds[buf][wn + tj * 16 + m][0];
      #pragma unroll
      for (int j = 0; j < 8; ++j)
        bf[tj].u[j] = *(const unsigned int*)&row[kh * 8 + (j >> 2) * 8 + 2 * j];
    }
    #pragma unroll
    for (int ti = 0; ti < 2; ++ti)
      #pragma unroll
      for (int tj = 0; tj < 2; ++tj)
        acc[ti][tj].v = wmma_bf16(a[ti], bf[tj], acc[ti][tj].v);
  }

  const int ncol = lane & 15, mh = (lane >> 4) * 8;
  #pragma unroll
  for (int ti = 0; ti < 2; ++ti)
    #pragma unroll
    for (int tj = 0; tj < 2; ++tj)
      #pragma unroll
      for (int i = 0; i < 8; ++i) {
        int orow = o0 + wo + ti * 16 + i + mh;
        int ng   = n0 + wn + tj * 16 + ncol;
        float val = (acc[ti][tj].f[i] + bias[orow]) * postscale;
        if (MODE == 0)
          ((float*)Out)[((size_t)b * C_DIM + orow) * N_DIM + ng] = val;
        else
          ((unsigned short*)Out)[((size_t)b * N_DIM + ng) * C_DIM + orow] = f2bf(val);
      }
}

// ---------------------------------------------------------------------------
// Flash attention: one wave owns 16 query positions of one (b,h).
// qT/kT bf16 [B][N][C] (q pre-scaled), vB bf16 [B][C][N]: all fragment
// sources are contiguous 4B loads -> b128-merged global loads feeding wmma.
// ---------------------------------------------------------------------------
__global__ __launch_bounds__(256) void attn_kernel(
    const unsigned short* __restrict__ qT, const unsigned short* __restrict__ kT,
    const unsigned short* __restrict__ vB, const int* __restrict__ mk,
    float* __restrict__ out) {
  alignas(16) __shared__ unsigned short Pb[8][16][KSTEP + 4];  // per-wave P^T, bf16
  __shared__ float RowB[8][16];
  const int lane = threadIdx.x & 31, wave = threadIdx.x >> 5;
  const int task = blockIdx.x * 8 + wave;
  const int b  = task >> 9;
  const int hh = (task >> 6) & 7;
  const int n0 = (task & 63) * 16;
  const unsigned short* vbb = vB + ((size_t)b * C_DIM + hh * HD) * N_DIM;
  const int*  mkb = mk + b * N_DIM;
  const int nl = lane & 15, half = lane >> 4;

  // q^T A-fragments (M=n, K=d): contiguous bf16 loads
  Frag aq[2];
  const int row_n = n0 + nl;
  const unsigned short* qrow = qT + ((size_t)b * N_DIM + row_n) * C_DIM + hh * HD;
  #pragma unroll
  for (int dblk = 0; dblk < 2; ++dblk)
    #pragma unroll
    for (int j = 0; j < 4; ++j) {
      aq[dblk].u[j]     = *(const unsigned int*)(qrow + dblk * 32 + half * 8 + 2 * j);
      aq[dblk].u[4 + j] = *(const unsigned int*)(qrow + dblk * 32 + 16 + half * 8 + 2 * j);
    }

  float mrow[8], lrow[8];
  Acc o_acc[4];
  #pragma unroll
  for (int i = 0; i < 8; ++i) { mrow[i] = -1e30f; lrow[i] = 0.f; }
  #pragma unroll
  for (int dt = 0; dt < 4; ++dt)
    #pragma unroll
    for (int i = 0; i < 8; ++i) o_acc[dt].f[i] = 0.f;

  for (int m0 = 0; m0 < N_DIM; m0 += 32) {
    // ---- S tiles (two 16n x 16m), K = 64 over d ----
    Acc s[2];
    #pragma unroll
    for (int msub = 0; msub < 2; ++msub) {
      const int mcol = m0 + msub * 16 + nl;
      const unsigned short* krow = kT + ((size_t)b * N_DIM + mcol) * C_DIM + hh * HD;
      const int koff = half * 8;
      Frag bk0, bk1;
      #pragma unroll
      for (int j = 0; j < 8; ++j) {
        bk0.u[j] = *(const unsigned int*)(krow + koff + (j >> 2) * 8 + 2 * j);
        bk1.u[j] = *(const unsigned int*)(krow + 32 + koff + (j >> 2) * 8 + 2 * j);
      }
      #pragma unroll
      for (int i = 0; i < 8; ++i) s[msub].f[i] = 0.f;
      s[msub].v = wmma_bf16(aq[0], bk0, s[msub].v);
      s[msub].v = wmma_bf16(aq[1], bk1, s[msub].v);
      int km = mkb[mcol];
      #pragma unroll
      for (int i = 0; i < 8; ++i)
        s[msub].f[i] = km ? s[msub].f[i] : -1e30f;
    }

    // ---- online softmax: rows indexed (vgpr i, lane-half) ----
    float alpha[8];
    #pragma unroll
    for (int i = 0; i < 8; ++i) {
      float cm = fmaxf(s[0].f[i], s[1].f[i]);
      for (int off = 1; off < 16; off <<= 1)
        cm = fmaxf(cm, __shfl_xor(cm, off, 32));
      float nm = fmaxf(mrow[i], cm);
      alpha[i] = __expf(mrow[i] - nm);
      mrow[i] = nm;
      s[0].f[i] = __expf(s[0].f[i] - nm);
      s[1].f[i] = __expf(s[1].f[i] - nm);
      float rs = s[0].f[i] + s[1].f[i];
      for (int off = 1; off < 16; off <<= 1)
        rs += __shfl_xor(rs, off, 32);
      lrow[i] = lrow[i] * alpha[i] + rs;
    }

    // ---- P -> per-wave LDS as bf16 (transpose), alpha broadcast ----
    #pragma unroll
    for (int msub = 0; msub < 2; ++msub)
      #pragma unroll
      for (int i = 0; i < 8; ++i)
        Pb[wave][i + 8 * half][msub * 16 + nl] = f2bf(s[msub].f[i]);
    if (nl == 0)
      #pragma unroll
      for (int i = 0; i < 8; ++i) RowB[wave][half * 8 + i] = alpha[i];
    float a_n = RowB[wave][nl];
    #pragma unroll
    for (int dt = 0; dt < 4; ++dt)
      #pragma unroll
      for (int i = 0; i < 8; ++i) o_acc[dt].f[i] *= a_n;

    // ---- O += v(16d x 32m) @ P^T(32m x 16n) ----
    Frag bp;
    const unsigned short* prow = &Pb[wave][nl][0];
    #pragma unroll
    for (int j = 0; j < 8; ++j)
      bp.u[j] = *(const unsigned int*)(prow + half * 8 + (j >> 2) * 8 + 2 * j);
    #pragma unroll
    for (int dt = 0; dt < 4; ++dt) {
      Frag av;
      const unsigned short* vrow = vbb + (size_t)(dt * 16 + nl) * N_DIM + m0;
      #pragma unroll
      for (int j = 0; j < 4; ++j) {
        av.u[j]     = *(const unsigned int*)(vrow + half * 8 + 2 * j);
        av.u[4 + j] = *(const unsigned int*)(vrow + 16 + half * 8 + 2 * j);
      }
      o_acc[dt].v = wmma_bf16(av, bp, o_acc[dt].v);
    }
  }

  // ---- finalize: divide by l, store [d, n] ----
  if (nl == 0)
    #pragma unroll
    for (int i = 0; i < 8; ++i) RowB[wave][half * 8 + i] = lrow[i];
  float linv = 1.0f / RowB[wave][nl];
  float* ob = out + ((size_t)b * C_DIM + hh * HD) * N_DIM;
  #pragma unroll
  for (int dt = 0; dt < 4; ++dt)
    #pragma unroll
    for (int i = 0; i < 8; ++i) {
      int drow = dt * 16 + i + 8 * half;
      ob[(size_t)drow * N_DIM + n0 + nl] = o_acc[dt].f[i] * linv;
    }
}

// ---------------------------------------------------------------------------
// v_newT[b,n,c] (bf16) = bf16( mq[b,n] ? attn_out[b,c,n] : v[b,c,n] )
// ---------------------------------------------------------------------------
__global__ __launch_bounds__(256) void scatter_kernel(
    const float* __restrict__ ov, const float* __restrict__ v,
    const int* __restrict__ mq, unsigned short* __restrict__ vnewT) {
  size_t idx = (size_t)blockIdx.x * 256 + threadIdx.x;
  int b = (int)(idx / ((size_t)C_DIM * N_DIM));
  int r = (int)(idx % ((size_t)C_DIM * N_DIM));
  int c = r / N_DIM, n = r % N_DIM;
  float val = mq[b * N_DIM + n] ? ov[idx] : v[idx];
  vnewT[((size_t)b * N_DIM + n) * C_DIM + c] = f2bf(val);
}

// ---------------------------------------------------------------------------
extern "C" void kernel_launch(void* const* d_in, const int* in_sizes, int n_in,
                              void* d_out, int out_size, void* d_ws, size_t ws_size,
                              hipStream_t stream) {
  const float* x    = (const float*)d_in[0];
  const float* u_q  = (const float*)d_in[1];
  const float* u_k  = (const float*)d_in[2];
  const float* wq_s = (const float*)d_in[3];
  const float* bq_s = (const float*)d_in[4];
  const float* wk_s = (const float*)d_in[5];
  const float* bk_s = (const float*)d_in[6];
  const float* Wq   = (const float*)d_in[7];
  const float* bq   = (const float*)d_in[8];
  const float* Wk   = (const float*)d_in[9];
  const float* bk   = (const float*)d_in[10];
  const float* Wv   = (const float*)d_in[11];
  const float* bv   = (const float*)d_in[12];
  const float* Wp   = (const float*)d_in[13];
  const float* bp   = (const float*)d_in[14];

  const size_t SZ  = (size_t)B_SZ * C_DIM * N_DIM;   // 4 Mi elements
  const size_t WSZ = (size_t)C_DIM * C_DIM;          // 256 Ki elements
  unsigned short* qT   = (unsigned short*)d_ws;
  unsigned short* kT   = qT + SZ;
  unsigned short* xT   = kT + SZ;
  unsigned short* vnT  = xT + SZ;
  unsigned short* vbf  = vnT + SZ;
  unsigned short* wqb  = vbf + SZ;
  unsigned short* wkb  = wqb + WSZ;
  unsigned short* wvb  = wkb + WSZ;
  unsigned short* wpb  = wvb + WSZ;
  float* vw  = (float*)(wpb + WSZ);
  float* ow  = vw + SZ;
  int*   mq  = (int*)(ow + SZ);
  int*   mkp = mq + B_SZ * N_DIM;

  dim3 gg(C_DIM / 128, N_DIM / 64, B_SZ);            // 4 x 16 x 8

  // one-time bf16 conversions
  cvt_kernel<<<(unsigned)(WSZ / 256), 256, 0, stream>>>(Wq, wqb);
  cvt_kernel<<<(unsigned)(WSZ / 256), 256, 0, stream>>>(Wk, wkb);
  cvt_kernel<<<(unsigned)(WSZ / 256), 256, 0, stream>>>(Wv, wvb);
  cvt_kernel<<<(unsigned)(WSZ / 256), 256, 0, stream>>>(Wp, wpb);
  cvt_xT_kernel<<<(unsigned)(SZ / 256), 256, 0, stream>>>(x, xT);

  mask_kernel<<<(B_SZ * N_DIM) / 256, 256, 0, stream>>>(x, wq_s, bq_s, u_q, mq);
  mask_kernel<<<(B_SZ * N_DIM) / 256, 256, 0, stream>>>(x, wk_s, bk_s, u_k, mkp);

  // q (bf16, transposed, pre-scaled by hd^-0.5), k (bf16, transposed), v (f32)
  gemm512b<1><<<gg, 256, 0, stream>>>(xT, wqb, bq, qT, 0.125f);
  gemm512b<1><<<gg, 256, 0, stream>>>(xT, wkb, bk, kT, 1.0f);
  gemm512b<0><<<gg, 256, 0, stream>>>(xT, wvb, bv, vw, 1.0f);

  // v -> bf16 [B][C][N] for the PV matmul A-fragments
  cvt_kernel<<<(unsigned)(SZ / 256), 256, 0, stream>>>(vw, vbf);

  attn_kernel<<<(B_SZ * HEADS * (N_DIM / 16)) / 8, 256, 0, stream>>>(qT, kT, vbf, mkp, ow);

  scatter_kernel<<<(unsigned)(SZ / 256), 256, 0, stream>>>(ow, vw, mq, vnT);

  gemm512b<0><<<gg, 256, 0, stream>>>(vnT, wpb, bp, (float*)d_out, 1.0f);
}
